// CrossGCL_20306605376059
// MI455X (gfx1250) — matrix operand order repaired and checked by reading, not compiled, not run on hardware
//
#include <hip/hip_runtime.h>
#include <hip/hip_bf16.h>

// ---------------------------------------------------------------------------
// CrossGCL (graph cross-attention GCL layer) for MI455X / gfx1250
//   edge MLP :  [E,256] x [256,128] -> SiLU -> [E,128] x [128,128] -> SiLU
//   scatter  :  segment_sum over row  (global f32 atomics)
//   node MLP :  [Nq,256] x [256,128] -> SiLU -> [Nq,128] x [128,128] + resid
// All GEMM work on v_wmma_f32_16x16x32_f16 (wave32).
//
// v2: features pre-converted to f16 once (kills per-edge v_cvt chains),
//     each wave owns 4 M-tiles x 2 N-tiles so every B-fragment pair feeds
//     8 independent WMMAs (hides load latency instead of load->wait->wmma).
// ---------------------------------------------------------------------------

#define NQ   20000
#define NKV  20000
#define NE   640000
#define DD   128
#define HH   128

typedef __attribute__((ext_vector_type(16))) _Float16 v16h;
typedef __attribute__((ext_vector_type(8)))  _Float16 v8h;
typedef __attribute__((ext_vector_type(4)))  _Float16 v4h;
typedef __attribute__((ext_vector_type(8)))  float    v8f;

__device__ __forceinline__ float silu_f(float x) {
    return x / (1.0f + __expf(-x));
}

__device__ __forceinline__ v16h cat8(v8h lo, v8h hi) {
    return __builtin_shufflevector(lo, hi,
        0, 1, 2, 3, 4, 5, 6, 7, 8, 9, 10, 11, 12, 13, 14, 15);
}

// A-fragment (16x32 f16) chunk loader: caller passes pointer already offset by
// row*128 + kb*32 + (lane>=16 ? 8 : 0); chunks at +0 and +16 per ISA A layout.
__device__ __forceinline__ v16h load_a_frag_f16(const _Float16* __restrict__ src) {
    const v8h h0 = *(const v8h*)src;
    const v8h h1 = *(const v8h*)(src + 16);
    return cat8(h0, h1);
}

// ---------------------------------------------------------------------------
// Weight repack: f32 row-major [K,128] -> fragment-ready f16 B tiles.
// B fragment for (kb,nb): lane<16 holds K{kb*32+0..7,+16..23}, lane>=16 K{+8..15,+24..31},
// N = nb*16 + (lane&15); one contiguous 32B load per lane.
// ---------------------------------------------------------------------------
__global__ void pack_weights_kernel(const float* __restrict__ We1,
                                    const float* __restrict__ We2,
                                    const float* __restrict__ Wn1,
                                    const float* __restrict__ Wn2,
                                    _Float16* __restrict__ p1,
                                    _Float16* __restrict__ p2,
                                    _Float16* __restrict__ pn1,
                                    _Float16* __restrict__ pn2) {
    int g = blockIdx.x * 256 + threadIdx.x;    // 98304 total elements
    const float* W;
    _Float16*    P;
    int idx;
    if (g < 32768)       { W = We1; P = p1;  idx = g;          }   // 256x128
    else if (g < 49152)  { W = We2; P = p2;  idx = g - 32768;  }   // 128x128
    else if (g < 81920)  { W = Wn1; P = pn1; idx = g - 49152;  }   // 256x128
    else                 { W = Wn2; P = pn2; idx = g - 81920;  }   // 128x128
    const int j    = idx & 15;
    const int lane = (idx >> 4) & 31;
    const int t    = idx >> 9;
    const int nb   = t & 7;
    const int kb   = t >> 3;
    const int k = kb * 32 + ((lane >> 4) & 1) * 8 + (j & 7) + (j >> 3) * 16;
    const int n = nb * 16 + (lane & 15);
    P[idx] = (_Float16)W[k * 128 + n];
}

__global__ void zero_kernel(float4* __restrict__ p, int n4) {
    int i = blockIdx.x * 256 + threadIdx.x;
    if (i < n4) p[i] = make_float4(0.f, 0.f, 0.f, 0.f);
}

// f32 -> f16 feature conversion (4 elements / thread)
__global__ void cvt_f16_kernel(const float* __restrict__ src,
                               _Float16* __restrict__ dst, int n4) {
    int i = blockIdx.x * 256 + threadIdx.x;
    if (i < n4) {
        const float4 f = ((const float4*)src)[i];
        v4h h;
        h[0] = (_Float16)f.x; h[1] = (_Float16)f.y;
        h[2] = (_Float16)f.z; h[3] = (_Float16)f.w;
        ((v4h*)dst)[i] = h;
    }
}

// ---------------------------------------------------------------------------
// Edge kernel: 256 threads = 8 waves, block covers 128 edges.
// Wave group (4 waves) covers 64 edges; each wave: 4 M-tiles x 2 N-tiles.
// ---------------------------------------------------------------------------
__global__ __launch_bounds__(256) void edge_kernel(
    const _Float16* __restrict__ hq16, const _Float16* __restrict__ hkv16,
    const int* __restrict__ eidx,
    const _Float16* __restrict__ p1, const float* __restrict__ be1,
    const _Float16* __restrict__ p2, const float* __restrict__ be2,
    float* __restrict__ mij_out, float* __restrict__ agg) {

    __shared__ _Float16 sh[2][64][HH];     // two 64x128 f16 hidden tiles (32 KB)

    const int tid  = threadIdx.x;
    const int wave = tid >> 5;
    const int lane = tid & 31;
    const int grp  = wave >> 2;            // edge group (0/1)
    const int wq   = wave & 3;             // which pair of N tiles
    const int m    = lane & 15;
    const int half = lane >> 4;
    const int ebase = blockIdx.x * 128 + grp * 64;
    const int nb0   = wq * 2;

    int r[4], c[4];
#pragma unroll
    for (int mt = 0; mt < 4; ++mt) {
        const int e = ebase + mt * 16 + m;
        r[mt] = eidx[e];
        c[mt] = eidx[NE + e];
    }

    // ---------------- layer 1:  [64,256] x [256,32] per wave ----------------
    v8f acc[4][2];
#pragma unroll
    for (int mt = 0; mt < 4; ++mt) { acc[mt][0] = (v8f){}; acc[mt][1] = (v8f){}; }

#pragma unroll
    for (int kb = 0; kb < 8; ++kb) {
        v16h a[4];
#pragma unroll
        for (int mt = 0; mt < 4; ++mt) {
            const _Float16* src = (kb < 4)
                ? hq16  + (size_t)r[mt] * DD + kb * 32 + half * 8
                : hkv16 + (size_t)c[mt] * DD + (kb - 4) * 32 + half * 8;
            a[mt] = load_a_frag_f16(src);
        }
        const v16h b0 = *(const v16h*)(p1 + (((kb * 8 + nb0    ) * 32 + lane) << 4));
        const v16h b1 = *(const v16h*)(p1 + (((kb * 8 + nb0 + 1) * 32 + lane) << 4));
#pragma unroll
        for (int mt = 0; mt < 4; ++mt) {
            acc[mt][0] = __builtin_amdgcn_wmma_f32_16x16x32_f16(
                false, a[mt], false, b0, (short)0, acc[mt][0], false, false);
            acc[mt][1] = __builtin_amdgcn_wmma_f32_16x16x32_f16(
                false, a[mt], false, b1, (short)0, acc[mt][1], false, false);
        }
    }

    // bias + SiLU -> shared hidden tile ([edge][K] f16, this wave's 32 columns)
#pragma unroll
    for (int j = 0; j < 2; ++j) {
        const int n = (nb0 + j) * 16 + m;
        const float bias = be1[n];
#pragma unroll
        for (int mt = 0; mt < 4; ++mt)
#pragma unroll
            for (int rr = 0; rr < 8; ++rr) {
                const float x = acc[mt][j][rr] + bias;
                sh[grp][mt * 16 + rr + half * 8][n] = (_Float16)silu_f(x);
            }
    }
    __syncthreads();

    // ---------------- layer 2:  [64,128] x [128,32] per wave ----------------
    v8f acc2[4][2];
#pragma unroll
    for (int mt = 0; mt < 4; ++mt) { acc2[mt][0] = (v8f){}; acc2[mt][1] = (v8f){}; }

#pragma unroll
    for (int kb = 0; kb < 4; ++kb) {
        v16h a[4];
#pragma unroll
        for (int mt = 0; mt < 4; ++mt)
            a[mt] = load_a_frag_f16(&sh[grp][mt * 16 + m][kb * 32 + half * 8]);
        const v16h b0 = *(const v16h*)(p2 + (((kb * 8 + nb0    ) * 32 + lane) << 4));
        const v16h b1 = *(const v16h*)(p2 + (((kb * 8 + nb0 + 1) * 32 + lane) << 4));
#pragma unroll
        for (int mt = 0; mt < 4; ++mt) {
            acc2[mt][0] = __builtin_amdgcn_wmma_f32_16x16x32_f16(
                false, a[mt], false, b0, (short)0, acc2[mt][0], false, false);
            acc2[mt][1] = __builtin_amdgcn_wmma_f32_16x16x32_f16(
                false, a[mt], false, b1, (short)0, acc2[mt][1], false, false);
        }
    }

    // epilogue: SiLU -> mij out + scatter-add into agg[row]
#pragma unroll
    for (int j = 0; j < 2; ++j) {
        const int n = (nb0 + j) * 16 + m;
        const float bias = be2[n];
#pragma unroll
        for (int mt = 0; mt < 4; ++mt)
#pragma unroll
            for (int rr = 0; rr < 8; ++rr) {
                const int e = ebase + mt * 16 + rr + half * 8;
                const float s = silu_f(acc2[mt][j][rr] + bias);
                mij_out[(size_t)e * HH + n] = s;
                atomicAdd(&agg[(size_t)eidx[e] * HH + n], s);
            }
    }
}

// ---------------------------------------------------------------------------
// Node kernel: one 16-row tile per wave, A = [h_q16 | agg16], residual out.
// ---------------------------------------------------------------------------
__global__ __launch_bounds__(256) void node_kernel(
    const _Float16* __restrict__ hq16, const _Float16* __restrict__ agg16,
    const float* __restrict__ h_q,
    const _Float16* __restrict__ pn1, const float* __restrict__ bn1,
    const _Float16* __restrict__ pn2, const float* __restrict__ bn2,
    float* __restrict__ out_hq) {

    __shared__ _Float16 lds[8][16][HH];

    const int tid  = threadIdx.x;
    const int wave = tid >> 5;
    const int lane = tid & 31;
    const int m    = lane & 15;
    const int half = lane >> 4;
    const int tile = (blockIdx.x * 8 + wave) * 16;
    if (tile >= NQ) return;                 // wave-uniform exit (EXEC all-1 for WMMA)

    const int node = tile + m;
    const _Float16* __restrict__ aq = hq16  + (size_t)node * DD;
    const _Float16* __restrict__ ag = agg16 + (size_t)node * HH;

    v8f acc[8];
#pragma unroll
    for (int nb = 0; nb < 8; ++nb) acc[nb] = (v8f){};

#pragma unroll
    for (int kb = 0; kb < 8; ++kb) {
        const _Float16* src = (kb < 4) ? (aq + kb * 32 + half * 8)
                                       : (ag + (kb - 4) * 32 + half * 8);
        const v16h a = load_a_frag_f16(src);
#pragma unroll
        for (int nb = 0; nb < 8; ++nb) {
            const v16h b = *(const v16h*)(pn1 + (((kb * 8 + nb) * 32 + lane) << 4));
            acc[nb] = __builtin_amdgcn_wmma_f32_16x16x32_f16(
                false, a, false, b, (short)0, acc[nb], false, false);
        }
    }

#pragma unroll
    for (int nb = 0; nb < 8; ++nb) {
        const float bias = bn1[nb * 16 + m];
#pragma unroll
        for (int rr = 0; rr < 8; ++rr) {
            const float x = acc[nb][rr] + bias;
            lds[wave][rr + half * 8][nb * 16 + m] = (_Float16)silu_f(x);
        }
    }

    v8f acc2[8];
#pragma unroll
    for (int nb = 0; nb < 8; ++nb) acc2[nb] = (v8f){};

#pragma unroll
    for (int kb = 0; kb < 4; ++kb) {
        const v16h a = load_a_frag_f16(&lds[wave][m][kb * 32 + half * 8]);
#pragma unroll
        for (int nb = 0; nb < 8; ++nb) {
            const v16h b = *(const v16h*)(pn2 + (((kb * 8 + nb) * 32 + lane) << 4));
            acc2[nb] = __builtin_amdgcn_wmma_f32_16x16x32_f16(
                false, a, false, b, (short)0, acc2[nb], false, false);
        }
    }

    // residual epilogue: h_q + mlp_out + bn2
#pragma unroll
    for (int nb = 0; nb < 8; ++nb) {
        const int n = nb * 16 + m;
        const float bias = bn2[n];
#pragma unroll
        for (int rr = 0; rr < 8; ++rr) {
            const int nd = tile + rr + half * 8;
            out_hq[(size_t)nd * DD + n] =
                h_q[(size_t)nd * DD + n] + acc2[nb][rr] + bias;
        }
    }
}

// ---------------------------------------------------------------------------
// Host-side launcher
// ---------------------------------------------------------------------------
extern "C" void kernel_launch(void* const* d_in, const int* in_sizes, int n_in,
                              void* d_out, int out_size, void* d_ws, size_t ws_size,
                              hipStream_t stream) {
    (void)in_sizes; (void)n_in; (void)out_size; (void)ws_size;

    const float* h_q  = (const float*)d_in[0];
    const float* h_kv = (const float*)d_in[1];
    const int*   eidx = (const int*)d_in[2];
    const float* We1  = (const float*)d_in[3];
    const float* be1  = (const float*)d_in[4];
    const float* We2  = (const float*)d_in[5];
    const float* be2  = (const float*)d_in[6];
    const float* Wn1  = (const float*)d_in[7];
    const float* bn1  = (const float*)d_in[8];
    const float* Wn2  = (const float*)d_in[9];
    const float* bn2  = (const float*)d_in[10];

    // workspace layout (4 KB aligned offsets)
    char* ws = (char*)d_ws;
    float*     agg   = (float*)ws;                       // 10,240,000 B
    _Float16*  hq16  = (_Float16*)(ws + 10240000);       //  5,120,000 B
    _Float16*  hkv16 = (_Float16*)(ws + 15360000);       //  5,120,000 B
    _Float16*  agg16 = (_Float16*)(ws + 20480000);       //  5,120,000 B
    _Float16*  p1    = (_Float16*)(ws + 25600000);       // 64 KB
    _Float16*  p2    = (_Float16*)(ws + 25600000 + 65536);
    _Float16*  pn1   = (_Float16*)(ws + 25600000 + 98304);
    _Float16*  pn2   = (_Float16*)(ws + 25600000 + 163840);

    float* out_hq  = (float*)d_out;                      // [NQ,128]
    float* out_mij = out_hq + (size_t)NQ * DD;           // [E,128]

    // 1) zero aggregation buffer, repack weights, pre-convert features to f16
    zero_kernel<<<2500, 256, 0, stream>>>((float4*)agg, (NQ * HH) / 4);
    pack_weights_kernel<<<384, 256, 0, stream>>>(We1, We2, Wn1, Wn2, p1, p2, pn1, pn2);
    cvt_f16_kernel<<<2500, 256, 0, stream>>>(h_q,  hq16,  (NQ  * DD) / 4);
    cvt_f16_kernel<<<2500, 256, 0, stream>>>(h_kv, hkv16, (NKV * DD) / 4);

    // 2) edge MLP + scatter-add   (640000 edges / 128 per block)
    edge_kernel<<<NE / 128, 256, 0, stream>>>(hq16, hkv16, eidx, p1, be1, p2, be2,
                                              out_mij, agg);

    // 3) convert aggregated messages to f16 for the node GEMM
    cvt_f16_kernel<<<2500, 256, 0, stream>>>(agg, agg16, (NQ * HH) / 4);

    // 4) node MLP + residual      (20000 nodes, 1250 wave-tiles)
    node_kernel<<<157, 256, 0, stream>>>(hq16, agg16, h_q, pn1, bn1, pn2, bn2, out_hq);
}
